// DrugDiseaseFusionHGT_26036091748707
// MI455X (gfx1250) — matrix operand order repaired and checked by reading, not compiled, not run on hardware
//
#include <hip/hip_runtime.h>
#include <hip/hip_bf16.h>
#include <math.h>

typedef __attribute__((ext_vector_type(16))) _Float16 v16h;
typedef __attribute__((ext_vector_type(8)))  float    v8f;

#define FLAG_PREGELU  1
#define FLAG_POSTGELU 2

__device__ __forceinline__ float geluf(float x) {
  return 0.5f * x * (1.0f + erff(x * 0.70710678118654752f));
}
__device__ __forceinline__ float sigmf(float x) {
  return 1.0f / (1.0f + expf(-x));
}
__device__ __forceinline__ void atomicMaxF(float* addr, float val) {
  unsigned int* ua = (unsigned int*)addr;
  unsigned int old = __float_as_uint(*addr);
  while (__uint_as_float(old) < val) {
    unsigned int prev = atomicCAS(ua, old, __float_as_uint(val));
    if (prev == old) break;
    old = prev;
  }
}

// ---------------------------------------------------------------------------
// WMMA GEMM: C[M,N] = act(A[M,K]) * B[K,N] + bias, one wave per 16x16 tile.
// A (f16 16x32, ISA 7.12.2): lane l -> M=m0+l; packed element i<8 -> K=8*half+i,
//   i>=8 -> K=16+8*half+(i-8): two contiguous 8-float runs -> 4x b128 loads.
// B (32x16): staged via LDS. Lane loads B row (k0+lane), 16 contiguous floats
//   (4x b128), then fragment element i reads Bs[(i<8 ? 8*half+i : 8+8*half+i)*16 + l].
// C/D: element r -> M = m0 + r + 8*half, N = n0 + l.
// ---------------------------------------------------------------------------
template <int FLAGS>
__global__ void k_gemm_wmma(const float* __restrict__ A, int lda,
                            const float* __restrict__ B, int ldb,
                            const float* __restrict__ bias,
                            float* __restrict__ C, int ldc, int K) {
  __shared__ float Bs[512];
  const int lane = threadIdx.x;
  const int l = lane & 15, half = lane >> 4;
  const int m0 = blockIdx.y << 4, n0 = blockIdx.x << 4;
  const float* Arow = A + (size_t)(m0 + l) * lda;
  v8f acc = {0.f, 0.f, 0.f, 0.f, 0.f, 0.f, 0.f, 0.f};
  for (int k0 = 0; k0 < K; k0 += 32) {
    const float* Brow = B + (size_t)(k0 + lane) * ldb + n0;
    float4 bv0 = *(const float4*)(Brow + 0);
    float4 bv1 = *(const float4*)(Brow + 4);
    float4 bv2 = *(const float4*)(Brow + 8);
    float4 bv3 = *(const float4*)(Brow + 12);
    const float* Ap0 = Arow + k0 + 8 * half;
    const float* Ap1 = Arow + k0 + 16 + 8 * half;
    float4 av0 = *(const float4*)(Ap0 + 0);
    float4 av1 = *(const float4*)(Ap0 + 4);
    float4 av2 = *(const float4*)(Ap1 + 0);
    float4 av3 = *(const float4*)(Ap1 + 4);
    if (k0 + 32 < K) {
      __builtin_prefetch(Arow + k0 + 32, 0, 1);
      __builtin_prefetch(B + (size_t)(k0 + 32 + lane) * ldb + n0, 0, 1);
    }
    __syncthreads();
    ((float4*)Bs)[lane * 4 + 0] = bv0;
    ((float4*)Bs)[lane * 4 + 1] = bv1;
    ((float4*)Bs)[lane * 4 + 2] = bv2;
    ((float4*)Bs)[lane * 4 + 3] = bv3;
    __syncthreads();
    float af[16] = {av0.x, av0.y, av0.z, av0.w, av1.x, av1.y, av1.z, av1.w,
                    av2.x, av2.y, av2.z, av2.w, av3.x, av3.y, av3.z, av3.w};
    v16h a, b;
#pragma unroll
    for (int i = 0; i < 16; ++i) {
      float x = af[i];
      if (FLAGS & FLAG_PREGELU) x = geluf(x);
      a[i] = (_Float16)x;
    }
#pragma unroll
    for (int i = 0; i < 16; ++i)
      b[i] = (_Float16)Bs[(i < 8 ? 8 * half + i : 8 + 8 * half + i) * 16 + l];
    acc = __builtin_amdgcn_wmma_f32_16x16x32_f16(false, a, false, b,
                                                 (short)0, acc, false, false);
  }
  float bs = bias ? bias[n0 + l] : 0.0f;
#pragma unroll
  for (int r = 0; r < 8; ++r) {
    float v = acc[r] + bs;
    if (FLAGS & FLAG_POSTGELU) v = geluf(v);
    C[(size_t)(m0 + r + 8 * half) * ldc + (n0 + l)] = v;
  }
}

__global__ void k_fill(float* __restrict__ p, float v, int n) {
  int i = blockIdx.x * blockDim.x + threadIdx.x;
  if (i < n) p[i] = v;
}

// ---------------- HGT edge kernels (HEADS=4, DH=32) ----------------
__global__ void k_hgt_logit(const int* __restrict__ src, const int* __restrict__ dst,
                            const float* __restrict__ kqv_dst,   // q at col 128
                            const float* __restrict__ kr,        // [Nsrc,128]
                            const float* __restrict__ prel,      // [4]
                            float* __restrict__ logits, int E) {
  int t = blockIdx.x * blockDim.x + threadIdx.x;
  if (t >= E * 4) return;
  int e = t >> 2, h = t & 3;
  const float* q = kqv_dst + (size_t)dst[e] * 384 + 128 + h * 32;
  const float* k = kr + (size_t)src[e] * 128 + h * 32;
  float acc = 0.f;
#pragma unroll 8
  for (int i = 0; i < 32; ++i) acc += q[i] * k[i];
  logits[t] = acc * prel[h] * 0.17677669529663687f;  // /sqrt(32)
}

__global__ void k_segmax(const int* __restrict__ dst, const float* __restrict__ logits,
                         float* __restrict__ smax, int E) {
  int t = blockIdx.x * blockDim.x + threadIdx.x;
  if (t >= E * 4) return;
  int e = t >> 2, h = t & 3;
  atomicMaxF(&smax[dst[e] * 4 + h], logits[t]);
}

__global__ void k_segexp(const int* __restrict__ dst, float* __restrict__ logits,
                         const float* __restrict__ smax, float* __restrict__ ssum, int E) {
  int t = blockIdx.x * blockDim.x + threadIdx.x;
  if (t >= E * 4) return;
  int e = t >> 2, h = t & 3;
  float w = expf(logits[t] - smax[dst[e] * 4 + h]);
  logits[t] = w;
  atomicAdd(&ssum[dst[e] * 4 + h], w);
}

__global__ void k_hgt_agg(const int* __restrict__ src, const int* __restrict__ dst,
                          const float* __restrict__ vr, const float* __restrict__ expw,
                          const float* __restrict__ ssum, float* __restrict__ agg, int E) {
  int t = blockIdx.x * blockDim.x + threadIdx.x;
  if (t >= E * 128) return;
  int e = t >> 7, j = t & 127, h = j >> 5;
  float coeff = expw[e * 4 + h] / (ssum[dst[e] * 4 + h] + 1e-16f);
  atomicAdd(&agg[(size_t)dst[e] * 128 + j], vr[(size_t)src[e] * 128 + j] * coeff);
}

// out = sk*o + (1-sk)*h (skip!=null) else o;  H = gelu(LN(out + H))
__global__ void k_blend_ln_gelu(const float* __restrict__ O, float* __restrict__ H,
                                const float* __restrict__ skipp,
                                const float* __restrict__ g, const float* __restrict__ bt) {
  int n = blockIdx.x, c = threadIdx.x;
  __shared__ float red[128];
  size_t i = (size_t)n * 128 + c;
  float h = H[i], o = O[i];
  float ov;
  if (skipp) { float sk = sigmf(skipp[0]); ov = sk * o + (1.f - sk) * h; }
  else ov = o;
  float t = ov + h;
  red[c] = t; __syncthreads();
  for (int s = 64; s > 0; s >>= 1) { if (c < s) red[c] += red[c + s]; __syncthreads(); }
  float mu = red[0] * (1.f / 128.f);
  __syncthreads();
  float dv = t - mu;
  red[c] = dv * dv; __syncthreads();
  for (int s = 64; s > 0; s >>= 1) { if (c < s) red[c] += red[c + s]; __syncthreads(); }
  float var = red[0] * (1.f / 128.f);
  H[i] = geluf(dv * rsqrtf(var + 1e-5f) * g[c] + bt[c]);
}

// ---------------- Sim (TransformerConv) kernels: 4 heads x 128 ----------------
__global__ void k_sim_logit(const int* __restrict__ src, const int* __restrict__ dst,
                            const float* __restrict__ ea,
                            const float* __restrict__ qb, const float* __restrict__ kb,
                            const float* __restrict__ ew, const float* __restrict__ eb,
                            float* __restrict__ logits, int E) {
  int t = blockIdx.x * blockDim.x + threadIdx.x;
  if (t >= E * 4) return;
  int e = t >> 2, h = t & 3;
  float a = ea[e];
  const float* q = qb + (size_t)dst[e] * 512 + h * 128;
  const float* k = kb + (size_t)src[e] * 512 + h * 128;
  const float* w = ew + h * 128;
  const float* b = eb + h * 128;
  float acc = 0.f;
#pragma unroll 8
  for (int i = 0; i < 128; ++i) acc += q[i] * (k[i] + a * w[i] + b[i]);
  logits[t] = acc * 0.08838834764831845f;  // /sqrt(128)
}

__global__ void k_sim_agg(const int* __restrict__ src, const int* __restrict__ dst,
                          const float* __restrict__ ea, const float* __restrict__ vb,
                          const float* __restrict__ ew, const float* __restrict__ eb,
                          const float* __restrict__ expw, const float* __restrict__ ssum,
                          float* __restrict__ agg, int E) {
  int t = blockIdx.x * blockDim.x + threadIdx.x;
  if (t >= E * 512) return;
  int e = t >> 9, j = t & 511, h = j >> 7;
  float coeff = expw[e * 4 + h] / (ssum[dst[e] * 4 + h] + 1e-16f);
  float val = vb[(size_t)src[e] * 512 + j] + ea[e] * ew[j] + eb[j];
  atomicAdd(&agg[(size_t)dst[e] * 512 + j], val * coeff);
}

__global__ void k_headmean(const float* __restrict__ agg, float* __restrict__ out, int N) {
  int t = blockIdx.x * blockDim.x + threadIdx.x;
  if (t >= N * 128) return;
  int n = t >> 7, c = t & 127;
  const float* a = agg + (size_t)n * 512;
  out[t] = 0.25f * (a[c] + a[128 + c] + a[256 + c] + a[384 + c]);
}

// beta = sigmoid([out,xr,out-xr] @ wb);  tb = beta*xr + (1-beta)*out
__global__ void k_betablend(const float* __restrict__ outb, const float* __restrict__ xrb,
                            const float* __restrict__ wb, float* __restrict__ tb) {
  int n = blockIdx.x, c = threadIdx.x;
  __shared__ float red[128];
  size_t i = (size_t)n * 128 + c;
  float ov = outb[i], xr = xrb[i];
  red[c] = ov * wb[c] + xr * wb[128 + c] + (ov - xr) * wb[256 + c];
  __syncthreads();
  for (int s = 64; s > 0; s >>= 1) { if (c < s) red[c] += red[c + s]; __syncthreads(); }
  float beta = sigmf(red[0]);
  tb[i] = beta * xr + (1.f - beta) * ov;
}

// ---------------- Fusion / pairs ----------------
__global__ void k_build_gi(const float* __restrict__ a, const float* __restrict__ b,
                           float* __restrict__ gi, int N) {
  int t = blockIdx.x * blockDim.x + threadIdx.x;
  if (t >= N * 512) return;
  int n = t >> 9, j = t & 511, c = j & 127, sel = j >> 7;
  float av = a[(size_t)n * 128 + c], bv = b[(size_t)n * 128 + c];
  float v = (sel == 0) ? av : (sel == 1) ? bv : (sel == 2) ? av * bv : fabsf(av - bv);
  gi[(size_t)n * 512 + j] = v;
}

__global__ void k_gateblend(const float* __restrict__ g2, const float* __restrict__ a,
                            const float* __restrict__ b, float* __restrict__ fused, int n) {
  int t = blockIdx.x * blockDim.x + threadIdx.x;
  if (t >= n) return;
  float g = sigmf(g2[t]);
  fused[t] = g * a[t] + (1.f - g) * b[t];
}

__global__ void k_build_pe(const float* __restrict__ fd, const float* __restrict__ fs,
                           const int* __restrict__ di, const int* __restrict__ si,
                           float* __restrict__ pe, int base, int CH) {
  int t = blockIdx.x * blockDim.x + threadIdx.x;
  if (t >= CH * 512) return;
  int p = t >> 9, j = t & 511, c = j & 127, sel = j >> 7;
  float de = fd[(size_t)di[base + p] * 128 + c];
  float se = fs[(size_t)si[base + p] * 128 + c];
  float v = (sel == 0) ? de : (sel == 1) ? se : (sel == 2) ? de * se : fabsf(de - se);
  pe[(size_t)p * 512 + j] = v;
}

__global__ void k_decfinal(const float* __restrict__ h, const float* __restrict__ w,
                           const float* __restrict__ b, float* __restrict__ out, int CH) {
  int p = blockIdx.x * blockDim.x + threadIdx.x;
  if (p >= CH) return;
  const float* r = h + (size_t)p * 128;
  float acc = 0.f;
#pragma unroll 8
  for (int c = 0; c < 128; ++c) acc += r[c] * w[c];
  out[p] = acc + b[0];
}

static inline int divup(int a, int b) { return (a + b - 1) / b; }

extern "C" void kernel_launch(void* const* d_in, const int* in_sizes, int n_in,
                              void* d_out, int out_size, void* d_ws, size_t ws_size,
                              hipStream_t stream) {
  (void)in_sizes; (void)out_size;
  if (n_in < 204) return;  // expects 15 data inputs + 189 flattened params

  auto PF = [&](int i) -> const float* { return (const float*)d_in[i]; };
  auto PI = [&](int i) -> const int*   { return (const int*)d_in[i]; };

  const int NNODE[3] = {6000, 6000, 10000};   // drug, disease, gene
  const int SORTI[3] = {1, 0, 2};             // sorted-dict index of each type
  const int XIDX[3]  = {0, 1, 2};
  const int INDIM[3] = {1024, 512, 256};

  struct ETI { int in; int E; int st; int dt; int rs; };
  const ETI ets[6] = {
    {3,  80000, 0, 1, 3},  // drug  treats        disease
    {4,  80000, 1, 0, 1},  // dis   treated_by    drug
    {5, 100000, 0, 2, 2},  // drug  targets       gene
    {6, 100000, 2, 0, 5},  // gene  targeted_by   drug
    {7, 100000, 1, 2, 0},  // dis   associates    gene
    {8, 100000, 2, 1, 4},  // gene  assoc_with    disease
  };

  // ---- workspace bump allocator (floats) ----
  float* W = (float*)d_ws;
  size_t off = 0;
  auto WS = [&](size_t n) -> float* {
    float* p = W + off; off += (n + 63) & ~(size_t)63; return p;
  };
  float* proj[3]; for (int t = 0; t < 3; ++t) proj[t] = WS((size_t)NNODE[t] * 128);
  float* H[3];    for (int t = 0; t < 3; ++t) H[t]    = WS((size_t)NNODE[t] * 128);
  float* KQV[3];  for (int t = 0; t < 3; ++t) KQV[t]  = WS((size_t)NNODE[t] * 384);
  float* krbuf = WS((size_t)10000 * 128);
  float* vrbuf = WS((size_t)10000 * 128);
  size_t lofs[7]; lofs[0] = 0;
  for (int r = 0; r < 6; ++r) lofs[r + 1] = lofs[r] + (size_t)ets[r].E * 4;
  float* elog = WS(lofs[6]);
  float* smax[3]; float* hsum[3]; float* agg[3];
  for (int t = 0; t < 3; ++t) {
    smax[t] = WS((size_t)NNODE[t] * 4);
    hsum[t] = WS((size_t)NNODE[t] * 4);
    agg[t]  = WS((size_t)NNODE[t] * 128);
  }
  float* obuf = WS((size_t)10000 * 128);
  float* Z[2]; Z[0] = WS((size_t)6000 * 128); Z[1] = WS((size_t)6000 * 128);
  float* qb = WS((size_t)6000 * 512);
  float* kb = WS((size_t)6000 * 512);
  float* vb = WS((size_t)6000 * 512);
  float* slog  = WS((size_t)120000 * 4);
  float* ssmax = WS((size_t)6000 * 4);
  float* sssum = WS((size_t)6000 * 4);
  float* sagg  = WS((size_t)6000 * 512);
  float* soutb = WS((size_t)6000 * 128);
  float* xrb   = WS((size_t)6000 * 128);
  float* tbuf  = WS((size_t)6000 * 128);
  float* gi = WS((size_t)6000 * 512);
  float* g1 = WS((size_t)6000 * 128);
  float* g2 = WS((size_t)6000 * 128);
  float* fused[2]; fused[0] = WS((size_t)6000 * 128); fused[1] = WS((size_t)6000 * 128);
  const int CH = 16384;
  float* pe  = WS((size_t)CH * 512);
  float* ph1 = WS((size_t)CH * 256);
  float* ph2 = WS((size_t)CH * 128);
  if (off * sizeof(float) > ws_size) return;

  auto gemm = [&](const float* A, int lda, const float* B, int ldb, const float* bias,
                  float* C, int ldc, int M, int N, int K, int flags) {
    dim3 g(N / 16, M / 16);
    if (flags & FLAG_PREGELU)
      k_gemm_wmma<FLAG_PREGELU><<<g, dim3(32), 0, stream>>>(A, lda, B, ldb, bias, C, ldc, K);
    else if (flags & FLAG_POSTGELU)
      k_gemm_wmma<FLAG_POSTGELU><<<g, dim3(32), 0, stream>>>(A, lda, B, ldb, bias, C, ldc, K);
    else
      k_gemm_wmma<0><<<g, dim3(32), 0, stream>>>(A, lda, B, ldb, bias, C, ldc, K);
  };
  auto fill = [&](float* p, float v, size_t n) {
    k_fill<<<divup((int)n, 256), 256, 0, stream>>>(p, v, (int)n);
  };

  // ---- Stage 0: input projections; h = proj ----
  for (int t = 0; t < 3; ++t) {
    int wi = 146 + 2 * SORTI[t];
    gemm(PF(XIDX[t]), INDIM[t], PF(wi), 128, PF(wi + 1), proj[t], 128,
         NNODE[t], 128, INDIM[t], 0);
    hipMemcpyAsync(H[t], proj[t], (size_t)NNODE[t] * 128 * sizeof(float),
                   hipMemcpyDeviceToDevice, stream);
  }

  // ---- Stage 1: 3 HGT layers ----
  for (int L = 0; L < 3; ++L) {
    int base = 29 + 39 * L;
    for (int t = 0; t < 3; ++t) {                       // kqv projections
      int wi = base + 2 * SORTI[t];
      gemm(H[t], 128, PF(wi), 384, PF(wi + 1), KQV[t], 384, NNODE[t], 384, 128, 0);
    }
    for (int t = 0; t < 3; ++t) {
      fill(smax[t], -3.0e38f, (size_t)NNODE[t] * 4);
      fill(hsum[t], 0.f, (size_t)NNODE[t] * 4);
      fill(agg[t], 0.f, (size_t)NNODE[t] * 128);
    }
    for (int r = 0; r < 6; ++r) {                       // kr + logits
      const ETI& e = ets[r];
      const int* esrc = PI(e.in); const int* edst = esrc + e.E;
      const float* arel = PF(base + 18 + 3 * e.rs);
      const float* prel = PF(base + 18 + 3 * e.rs + 2);
      for (int hh = 0; hh < 4; ++hh)
        gemm(KQV[e.st] + hh * 32, 384, arel + hh * 1024, 32, nullptr,
             krbuf + hh * 32, 128, NNODE[e.st], 32, 32, 0);
      k_hgt_logit<<<divup(e.E * 4, 256), 256, 0, stream>>>(
          esrc, edst, KQV[e.dt], krbuf, prel, elog + lofs[r], e.E);
    }
    for (int r = 0; r < 6; ++r) {
      const ETI& e = ets[r];
      k_segmax<<<divup(e.E * 4, 256), 256, 0, stream>>>(
          PI(e.in) + e.E, elog + lofs[r], smax[e.dt], e.E);
    }
    for (int r = 0; r < 6; ++r) {
      const ETI& e = ets[r];
      k_segexp<<<divup(e.E * 4, 256), 256, 0, stream>>>(
          PI(e.in) + e.E, elog + lofs[r], smax[e.dt], hsum[e.dt], e.E);
    }
    for (int r = 0; r < 6; ++r) {                       // vr + weighted scatter
      const ETI& e = ets[r];
      const int* esrc = PI(e.in); const int* edst = esrc + e.E;
      const float* mrel = PF(base + 18 + 3 * e.rs + 1);
      for (int hh = 0; hh < 4; ++hh)
        gemm(KQV[e.st] + 256 + hh * 32, 384, mrel + hh * 1024, 32, nullptr,
             vrbuf + hh * 32, 128, NNODE[e.st], 32, 32, 0);
      k_hgt_agg<<<divup(e.E * 128, 256), 256, 0, stream>>>(
          esrc, edst, vrbuf, elog + lofs[r], hsum[e.dt], agg[e.dt], e.E);
    }
    for (int t = 0; t < 3; ++t) {                       // out proj + skip + LN + gelu
      int oi = base + 12 + 2 * SORTI[t];
      gemm(agg[t], 128, PF(oi), 128, PF(oi + 1), obuf, 128,
           NNODE[t], 128, 128, FLAG_PREGELU);
      k_blend_ln_gelu<<<NNODE[t], 128, 0, stream>>>(
          obuf, H[t], PF(base + 36 + SORTI[t]),
          PF(base + 6 + 2 * SORTI[t]), PF(base + 6 + 2 * SORTI[t] + 1));
    }
  }

  // ---- Stage 2: similarity TransformerConv stacks (drug, disease) ----
  for (int s = 0; s < 2; ++s) {
    int t = (s == 0) ? 0 : 1;                 // drug then disease
    int pb0 = (t == 0) ? 178 : 152;           // sorted: disease first at 152
    const int* ei = PI(t == 0 ? 9 : 11);
    const float* ea = PF(t == 0 ? 10 : 12);
    const int* esrc = ei; const int* edst = ei + 120000;
    float* z = Z[s];
    hipMemcpyAsync(z, proj[t], (size_t)6000 * 128 * sizeof(float),
                   hipMemcpyDeviceToDevice, stream);
    for (int L = 0; L < 2; ++L) {
      int pb = pb0 + 13 * L;  // beta,e_w,e_b,k_w,k_b,ln_g,ln_b,q_w,q_b,skip_w,skip_b,v_w,v_b
      gemm(z, 128, PF(pb + 7), 512, PF(pb + 8), qb, 512, 6000, 512, 128, 0);
      gemm(z, 128, PF(pb + 3), 512, PF(pb + 4), kb, 512, 6000, 512, 128, 0);
      gemm(z, 128, PF(pb + 11), 512, PF(pb + 12), vb, 512, 6000, 512, 128, 0);
      fill(ssmax, -3.0e38f, 6000 * 4);
      fill(sssum, 0.f, 6000 * 4);
      fill(sagg, 0.f, (size_t)6000 * 512);
      k_sim_logit<<<divup(120000 * 4, 256), 256, 0, stream>>>(
          esrc, edst, ea, qb, kb, PF(pb + 1), PF(pb + 2), slog, 120000);
      k_segmax<<<divup(120000 * 4, 256), 256, 0, stream>>>(edst, slog, ssmax, 120000);
      k_segexp<<<divup(120000 * 4, 256), 256, 0, stream>>>(edst, slog, ssmax, sssum, 120000);
      k_sim_agg<<<divup(120000 * 512, 256), 256, 0, stream>>>(
          esrc, edst, ea, vb, PF(pb + 1), PF(pb + 2), slog, sssum, sagg, 120000);
      k_headmean<<<divup(6000 * 128, 256), 256, 0, stream>>>(sagg, soutb, 6000);
      gemm(z, 128, PF(pb + 9), 128, PF(pb + 10), xrb, 128, 6000, 128, 128, 0);
      k_betablend<<<6000, 128, 0, stream>>>(soutb, xrb, PF(pb + 0), tbuf);
      k_blend_ln_gelu<<<6000, 128, 0, stream>>>(tbuf, z, nullptr,
                                                PF(pb + 5), PF(pb + 6));
    }
  }

  // ---- Stage 3: gated fusion ----
  for (int s = 0; s < 2; ++s) {
    int t = (s == 0) ? 0 : 1;
    int gb = (t == 0) ? 25 : 21;   // sorted: disease gate first at 21
    k_build_gi<<<divup(6000 * 512, 256), 256, 0, stream>>>(H[t], Z[s], gi, 6000);
    gemm(gi, 512, PF(gb), 128, PF(gb + 1), g1, 128, 6000, 128, 512, FLAG_POSTGELU);
    gemm(g1, 128, PF(gb + 2), 128, PF(gb + 3), g2, 128, 6000, 128, 128, 0);
    k_gateblend<<<divup(6000 * 128, 256), 256, 0, stream>>>(
        g2, H[t], Z[s], fused[s], 6000 * 128);
  }

  // ---- Stage 4: pair decoder (chunked) ----
  const int* dix = PI(13); const int* six = PI(14);
  float* outp = (float*)d_out;
  for (int c0 = 0; c0 < 65536; c0 += CH) {
    k_build_pe<<<divup(CH * 512, 256), 256, 0, stream>>>(
        fused[0], fused[1], dix, six, pe, c0, CH);
    gemm(pe, 512, PF(15), 256, PF(16), ph1, 256, CH, 256, 512, FLAG_POSTGELU);
    gemm(ph1, 256, PF(17), 128, PF(18), ph2, 128, CH, 128, 256, FLAG_POSTGELU);
    k_decfinal<<<divup(CH, 256), 256, 0, stream>>>(ph2, PF(19), PF(20), outp + c0, CH);
  }
}
